// SparseAttention_32667521253633
// MI455X (gfx1250) — matrix operand order repaired and checked by reading, not compiled, not run on hardware
//
#include <hip/hip_runtime.h>

typedef unsigned short u16;
typedef __attribute__((ext_vector_type(16))) __bf16 v16bf;
typedef __attribute__((ext_vector_type(8)))  float  v8f;
typedef __attribute__((ext_vector_type(4)))  unsigned int u32x4;

#define L_SEQ   2048
#define D_MODEL 1024
#define N_HEADS 16
#define HEAD_DIM 64
#define NG      204          // n_global = int(2048*0.1)
#define GP      224          // padded global count (13 tiles of 16 + AV pad)
#define NEG_INF (-1000000000.0f)

union Frag { v16bf v; u32x4 q[2]; };

__device__ __forceinline__ u16 f2bf(float f) {
    unsigned u = __builtin_bit_cast(unsigned, f);
    unsigned r = u + 0x7FFFu + ((u >> 16) & 1u);
    return (u16)(r >> 16);
}

__device__ __forceinline__ v8f wmma_bf16(const Frag& a, const Frag& b, v8f c) {
    return __builtin_amdgcn_wmma_f32_16x16x32_bf16(false, a.v, false, b.v,
                                                   (short)0, c, false, false);
}

__device__ __forceinline__ float hred_max(float v) {
    #pragma unroll
    for (int o = 1; o < 16; o <<= 1) v = fmaxf(v, __shfl_xor(v, o, 32));
    return v;
}
__device__ __forceinline__ float hred_sum(float v) {
    #pragma unroll
    for (int o = 1; o < 16; o <<= 1) v += __shfl_xor(v, o, 32);
    return v;
}

// async global -> LDS (CDNA5): VDST = LDS byte address, GVS addressing
__device__ __forceinline__ void async_ld128(unsigned lds_byte, const void* sbase, int vbyte) {
    asm volatile("global_load_async_to_lds_b128 %0, %1, %2 offset:0"
                 :: "v"(lds_byte), "v"(vbyte), "s"(sbase) : "memory");
}
__device__ __forceinline__ void wait_async_le5() {
    asm volatile("s_wait_asynccnt 5" ::: "memory");
}
__device__ __forceinline__ void wait_async_0() {
    asm volatile("s_wait_asynccnt 0" ::: "memory");
}

// ---------------- elementwise fp32 -> bf16 ----------------
__global__ void cvt_f32_bf16(const float* __restrict__ s, u16* __restrict__ d, int n) {
    int i = blockIdx.x * blockDim.x + threadIdx.x;
    if (i < n) d[i] = f2bf(s[i]);
}

// ---------------- aux: V transpose, global gather ----------------
__global__ void build_aux(const u16* __restrict__ Kb, const u16* __restrict__ Vb,
                          const int* __restrict__ gi,
                          u16* __restrict__ Vt,   // [D_MODEL][L_SEQ]
                          u16* __restrict__ Kg,   // [GP][D_MODEL]
                          u16* __restrict__ Vg) { // [D_MODEL][GP]
    int i = blockIdx.x * blockDim.x + threadIdx.x;
    if (i < D_MODEL * L_SEQ) {
        int d = i / L_SEQ, l = i % L_SEQ;
        Vt[i] = Vb[(size_t)l * D_MODEL + d];
    }
    if (i < GP * D_MODEL) {
        int g = i / D_MODEL, c = i % D_MODEL;
        Kg[i] = (g < NG) ? Kb[(size_t)gi[g] * D_MODEL + c] : (u16)0;
    }
    if (i < D_MODEL * GP) {
        int d = i / GP, g = i % GP;
        Vg[i] = (g < NG) ? Vb[(size_t)gi[g] * D_MODEL + d] : (u16)0;
    }
}

// ---------------- WMMA GEMM with double-buffered async LDS staging ----------------
// block = 4 waves -> 32 rows x 128 cols; per K-slab (32): stage X(32x32) + W(128x32)
// into LDS with global_load_async_to_lds_b128, consume with WMMA from LDS fragments.
#define SX_ELEMS 1024                    // 32*32 per buffer
#define SW_ELEMS 4096                    // 128*32 per buffer
#define SW_BASE  (2 * SX_ELEMS)          // element offset of W region in arena

template <bool OUT_BF16>
__global__ __launch_bounds__(128) void gemm_bias_kernel(const u16* __restrict__ A,
                                                        const u16* __restrict__ W,
                                                        const float* __restrict__ bias,
                                                        void* __restrict__ outp) {
    __shared__ __align__(16) u16 SMEM[2 * SX_ELEMS + 2 * SW_ELEMS];   // 20KB, LDS offset 0

    const int t    = threadIdx.x;
    const int lane = t & 31;
    const int w    = t >> 5;                       // wave in block 0..3
    const int m0   = (blockIdx.x & 63) * 32;       // 64 row blocks
    const int n0   = (blockIdx.x >> 6) * 128;      // 8 col blocks
    const int mh   = w >> 1;                       // wave row half
    const int nh   = w & 1;                        // wave col half
    const int n    = lane & 15;
    const int hl   = lane >> 4;

    const int xrow = t >> 2, cg = (t & 3) * 8;     // staging coords (8 bf16 = 16B)

    // issue one K-slab stage: 1 X load + 4 W loads per thread (all 128 threads)
    auto issue = [&](int kc, int buf) {
        const u16* ax = A + (size_t)m0 * D_MODEL + kc * 32;
        async_ld128((unsigned)((buf * SX_ELEMS + xrow * 32 + cg) * 2),
                    (const void*)ax, xrow * (D_MODEL * 2) + cg * 2);
        const u16* aw = W + (size_t)n0 * D_MODEL + kc * 32;
        #pragma unroll
        for (int j = 0; j < 4; ++j) {
            int wrow = j * 32 + xrow;
            async_ld128((unsigned)((SW_BASE + buf * SW_ELEMS + wrow * 32 + cg) * 2),
                        (const void*)aw, wrow * (D_MODEL * 2) + cg * 2);
        }
    };

    v8f acc[4] = {v8f{}, v8f{}, v8f{}, v8f{}};
    issue(0, 0);

    for (int kc = 0; kc < D_MODEL / 32; ++kc) {
        const int cur = kc & 1;
        const bool more = (kc + 1) < (D_MODEL / 32);
        if (more) issue(kc + 1, cur ^ 1);
        if (more) wait_async_le5(); else wait_async_0();
        __syncthreads();

        const u16* xs = &SMEM[cur * SX_ELEMS];
        const u16* ws = &SMEM[SW_BASE + cur * SW_ELEMS];
        Frag a;
        const u16* arow = xs + (mh * 16 + n) * 32;
        a.q[0] = *(const u32x4*)(arow + hl * 16);
        a.q[1] = *(const u32x4*)(arow + hl * 16 + 8);
        #pragma unroll
        for (int tt = 0; tt < 4; ++tt) {
            const u16* wrow = ws + (nh * 64 + tt * 16 + n) * 32 + hl * 16;
            Frag b;
            b.q[0] = *(const u32x4*)(wrow);
            b.q[1] = *(const u32x4*)(wrow + 8);
            acc[tt] = wmma_bf16(a, b, acc[tt]);
        }
        __syncthreads();
    }

    const int m0w = m0 + mh * 16;
    const int n0w = n0 + nh * 64;
    #pragma unroll
    for (int tt = 0; tt < 4; ++tt) {
        float bv = bias[n0w + tt * 16 + n];
        #pragma unroll
        for (int r = 0; r < 8; ++r) {
            int m = r + 8 * hl;
            float v = acc[tt][r] + bv;
            size_t idx = (size_t)(m0w + m) * D_MODEL + n0w + tt * 16 + n;
            if (OUT_BF16) ((u16*)outp)[idx] = f2bf(v);
            else          ((float*)outp)[idx] = v;
        }
    }
}

// ---------------- attention: one wave per (head, 16-query block) ----------------
__global__ __launch_bounds__(128) void attn_kernel(const u16* __restrict__ Qb,
                                                   const u16* __restrict__ Kb,
                                                   const u16* __restrict__ Vt,
                                                   const u16* __restrict__ Kg,
                                                   const u16* __restrict__ Vg,
                                                   u16* __restrict__ Ab) {
    __shared__ __align__(16) u16 P[4][16 * GP];     // per-wave prob tile (zero padded)
    const int lane  = threadIdx.x & 31;
    const int wslot = threadIdx.x >> 5;
    const int wid   = blockIdx.x * 4 + wslot;
    const int h     = wid >> 7;                     // 0..15
    const int qb    = wid & 127;                    // 0..127
    const int l0    = qb * 16;
    const int n     = lane & 15;
    const int hl    = lane >> 4;
    u16* pw = &P[wslot][0];

    for (int i = lane; i < 16 * GP; i += 32) pw[i] = 0;
    __syncthreads();

    // ---- Q A-fragments (Dh=64 -> two K=32 chunks), rows l0..l0+15 ----
    Frag aq[2];
    const u16* qrow = Qb + (size_t)(l0 + n) * D_MODEL + h * HEAD_DIM;
    #pragma unroll
    for (int kc = 0; kc < 2; ++kc) {
        aq[kc].q[0] = *(const u32x4*)(qrow + kc * 32 + hl * 16);
        aq[kc].q[1] = *(const u32x4*)(qrow + kc * 32 + hl * 16 + 8);
    }

    // prefetch V data used by the AV stages (global_prefetch_b8)
    __builtin_prefetch(Vt + (size_t)(h * HEAD_DIM + n) * L_SEQ + l0, 0, 3);
    __builtin_prefetch(Vg + (size_t)(h * HEAD_DIM + n) * GP, 0, 3);

    // ======== LOCAL: keys l0-32 .. l0+47 (80, padded to 96) ========
    const int keybase = l0 - 32;
    v8f sl[5];
    #pragma unroll
    for (int t = 0; t < 5; ++t) {
        v8f acc = v8f{};
        int key  = keybase + t * 16 + n;
        int keyc = min(max(key, 0), L_SEQ - 1);
        const u16* krow = Kb + (size_t)keyc * D_MODEL + h * HEAD_DIM;
        #pragma unroll
        for (int kc = 0; kc < 2; ++kc) {
            Frag b;
            b.q[0] = *(const u32x4*)(krow + kc * 32 + hl * 16);
            b.q[1] = *(const u32x4*)(krow + kc * 32 + hl * 16 + 8);
            acc = wmma_bf16(aq[kc], b, acc);
        }
        int j = t * 16 + n;
        bool keyok = (key >= 0) && (key < L_SEQ);
        #pragma unroll
        for (int r = 0; r < 8; ++r) {
            int m = r + 8 * hl;
            bool ok = keyok && (j >= m) && (j <= m + 64);
            acc[r] = ok ? acc[r] * 0.125f : NEG_INF;
        }
        sl[t] = acc;
    }
    // softmax (row m lives in reg r across the 16 lanes of half hl)
    #pragma unroll
    for (int r = 0; r < 8; ++r) {
        float mx = sl[0][r];
        #pragma unroll
        for (int t = 1; t < 5; ++t) mx = fmaxf(mx, sl[t][r]);
        mx = hred_max(mx);
        float sum = 0.f;
        #pragma unroll
        for (int t = 0; t < 5; ++t) { float p = __expf(sl[t][r] - mx); sl[t][r] = p; sum += p; }
        sum = hred_sum(sum);
        float inv = 1.0f / sum;
        int m = r + 8 * hl;
        #pragma unroll
        for (int t = 0; t < 5; ++t) pw[m * GP + t * 16 + n] = f2bf(sl[t][r] * inv);
    }
    __syncthreads();

    // local AV: P(16x96) @ Vrange(96x64)
    v8f lout[4];
    #pragma unroll
    for (int td = 0; td < 4; ++td) {
        v8f acc = v8f{};
        const u16* vrow = Vt + (size_t)(h * HEAD_DIM + td * 16 + n) * L_SEQ;
        const u16* prow = pw + n * GP;
        #pragma unroll
        for (int kc = 0; kc < 3; ++kc) {
            Frag a, b;
            a.q[0] = *(const u32x4*)(prow + kc * 32 + hl * 16);
            a.q[1] = *(const u32x4*)(prow + kc * 32 + hl * 16 + 8);
            int st = keybase + kc * 32 + hl * 16;
            st = min(max(st, 0), L_SEQ - 16);      // clamped runs have all-zero P
            b.q[0] = *(const u32x4*)(vrow + st);
            b.q[1] = *(const u32x4*)(vrow + st + 8);
            acc = wmma_bf16(a, b, acc);
        }
        lout[td] = acc;
    }
    __syncthreads();

    // ======== GLOBAL: 204 keys padded to 208 (13 tiles) ========
    v8f sg[13];
    #pragma unroll
    for (int t = 0; t < 13; ++t) {
        v8f acc = v8f{};
        const u16* krow = Kg + (size_t)(t * 16 + n) * D_MODEL + h * HEAD_DIM;
        #pragma unroll
        for (int kc = 0; kc < 2; ++kc) {
            Frag b;
            b.q[0] = *(const u32x4*)(krow + kc * 32 + hl * 16);
            b.q[1] = *(const u32x4*)(krow + kc * 32 + hl * 16 + 8);
            acc = wmma_bf16(aq[kc], b, acc);
        }
        bool ok = (t * 16 + n) < NG;
        #pragma unroll
        for (int r = 0; r < 8; ++r) acc[r] = ok ? acc[r] * 0.125f : NEG_INF;
        sg[t] = acc;
    }
    #pragma unroll
    for (int r = 0; r < 8; ++r) {
        float mx = sg[0][r];
        #pragma unroll
        for (int t = 1; t < 13; ++t) mx = fmaxf(mx, sg[t][r]);
        mx = hred_max(mx);
        float sum = 0.f;
        #pragma unroll
        for (int t = 0; t < 13; ++t) { float p = __expf(sg[t][r] - mx); sg[t][r] = p; sum += p; }
        sum = hred_sum(sum);
        float inv = 1.0f / sum;
        int m = r + 8 * hl;
        #pragma unroll
        for (int t = 0; t < 13; ++t) pw[m * GP + t * 16 + n] = f2bf(sg[t][r] * inv);
    }
    __syncthreads();

    // global AV: P(16x224) @ Vg(224x64)
    v8f gout[4];
    #pragma unroll
    for (int td = 0; td < 4; ++td) {
        v8f acc = v8f{};
        const u16* vrow = Vg + (size_t)(h * HEAD_DIM + td * 16 + n) * GP;
        const u16* prow = pw + n * GP;
        #pragma unroll
        for (int kc = 0; kc < 7; ++kc) {
            Frag a, b;
            a.q[0] = *(const u32x4*)(prow + kc * 32 + hl * 16);
            a.q[1] = *(const u32x4*)(prow + kc * 32 + hl * 16 + 8);
            b.q[0] = *(const u32x4*)(vrow + kc * 32 + hl * 16);
            b.q[1] = *(const u32x4*)(vrow + kc * 32 + hl * 16 + 8);
            acc = wmma_bf16(a, b, acc);
        }
        gout[td] = acc;
    }

    // blend 0.7*local + 0.3*global -> bf16
    #pragma unroll
    for (int td = 0; td < 4; ++td) {
        #pragma unroll
        for (int r = 0; r < 8; ++r) {
            int m = r + 8 * hl;
            float v = 0.7f * lout[td][r] + 0.3f * gout[td][r];
            Ab[(size_t)(l0 + m) * D_MODEL + h * HEAD_DIM + td * 16 + n] = f2bf(v);
        }
    }
}

// ---------------- host launch ----------------
extern "C" void kernel_launch(void* const* d_in, const int* in_sizes, int n_in,
                              void* d_out, int out_size, void* d_ws, size_t ws_size,
                              hipStream_t stream) {
    const float* x  = (const float*)d_in[0];
    const float* Wq = (const float*)d_in[1];
    const float* bq = (const float*)d_in[2];
    const float* Wk = (const float*)d_in[3];
    const float* bk = (const float*)d_in[4];
    const float* Wv = (const float*)d_in[5];
    const float* bv = (const float*)d_in[6];
    const float* Wo = (const float*)d_in[7];
    const float* bo = (const float*)d_in[8];
    const int*   gi = (const int*)d_in[9];

    char* p = (char*)d_ws;
    auto alloc = [&](size_t bytes) -> void* {
        void* r = (void*)p;
        p += (bytes + 255) & ~(size_t)255;
        return r;
    };
    u16* Xb  = (u16*)alloc((size_t)L_SEQ * D_MODEL * 2);
    u16* Wqb = (u16*)alloc((size_t)D_MODEL * D_MODEL * 2);
    u16* Wkb = (u16*)alloc((size_t)D_MODEL * D_MODEL * 2);
    u16* Wvb = (u16*)alloc((size_t)D_MODEL * D_MODEL * 2);
    u16* Wob = (u16*)alloc((size_t)D_MODEL * D_MODEL * 2);
    u16* Qb  = (u16*)alloc((size_t)L_SEQ * D_MODEL * 2);
    u16* Kb  = (u16*)alloc((size_t)L_SEQ * D_MODEL * 2);
    u16* Vb  = (u16*)alloc((size_t)L_SEQ * D_MODEL * 2);
    u16* Vt  = (u16*)alloc((size_t)D_MODEL * L_SEQ * 2);
    u16* Kg  = (u16*)alloc((size_t)GP * D_MODEL * 2);
    u16* Vg  = (u16*)alloc((size_t)D_MODEL * GP * 2);
    u16* Ab  = (u16*)alloc((size_t)L_SEQ * D_MODEL * 2);

    const int NXD = L_SEQ * D_MODEL;      // 2M
    const int NWW = D_MODEL * D_MODEL;    // 1M
    cvt_f32_bf16<<<(NXD + 255) / 256, 256, 0, stream>>>(x,  Xb,  NXD);
    cvt_f32_bf16<<<(NWW + 255) / 256, 256, 0, stream>>>(Wq, Wqb, NWW);
    cvt_f32_bf16<<<(NWW + 255) / 256, 256, 0, stream>>>(Wk, Wkb, NWW);
    cvt_f32_bf16<<<(NWW + 255) / 256, 256, 0, stream>>>(Wv, Wvb, NWW);
    cvt_f32_bf16<<<(NWW + 255) / 256, 256, 0, stream>>>(Wo, Wob, NWW);

    // 64 row-blocks * 8 col-blocks = 512 blocks, 4 waves each (32x128 tile/block)
    gemm_bias_kernel<true><<<512, 128, 0, stream>>>(Xb, Wqb, bq, (void*)Qb);
    gemm_bias_kernel<true><<<512, 128, 0, stream>>>(Xb, Wkb, bk, (void*)Kb);
    gemm_bias_kernel<true><<<512, 128, 0, stream>>>(Xb, Wvb, bv, (void*)Vb);

    build_aux<<<(NXD + 255) / 256, 256, 0, stream>>>(Kb, Vb, gi, Vt, Kg, Vg);

    // 16 heads * 128 query blocks = 2048 waves, 4 waves/block
    attn_kernel<<<512, 128, 0, stream>>>(Qb, Kb, Vt, Kg, Vg, Ab);

    gemm_bias_kernel<false><<<512, 128, 0, stream>>>(Ab, Wob, bo, d_out);
}
